// ScaledDotProductAttention_33784212751011
// MI455X (gfx1250) — matrix operand order repaired and checked by reading, not compiled, not run on hardware
//
#include <hip/hip_runtime.h>
#include <cstdint>
#include <cstddef>

// ---------------------------------------------------------------------------
// Scaled dot-product attention for MI455X (gfx1250), wave32 + WMMA bf16.
//   B=16, S=4096, E=768, D=64.
// All three GEMMs (QKV proj, Q.K^T, P.V) on v_wmma_f32_16x16x32_bf16 with
// f32 accumulation. Attention computed transposed (S^T = K x Q^T,
// O^T = V^T x P^T) so softmax stats are one scalar per lane and the P
// re-layout is a single shfl_xor(16) half-exchange.
// K/V tiles are staged block-wide into LDS with the CDNA5 async global->LDS
// data mover (ASYNCcnt + triple buffering): each tile fetched once per
// workgroup (8x traffic cut), copy of tile t+1 overlaps WMMA/softmax of t.
// Round-3 fix: async builtin takes pointers to GCC-vector int4 in the
// device/shared address spaces — cast accordingly.
// ---------------------------------------------------------------------------

#define EMBED  768
#define DQKV   64
#define SEQ    4096
#define NB     16

#if __has_builtin(__builtin_amdgcn_global_load_async_to_lds_b128) && \
    __has_builtin(__builtin_amdgcn_s_wait_asynccnt)
#define HAVE_ASYNC_LDS 1
#else
#define HAVE_ASYNC_LDS 0
#endif

typedef __bf16 bf16_t;
typedef bf16_t        v16bf __attribute__((ext_vector_type(16)));
typedef float         v8f   __attribute__((ext_vector_type(8)));
typedef unsigned int  u32x4 __attribute__((ext_vector_type(4)));
typedef float         f32x4 __attribute__((ext_vector_type(4)));
typedef unsigned short u16x4 __attribute__((ext_vector_type(4)));

#if HAVE_ASYNC_LDS
// Types matching the async builtin's parameters:
//   'int __attribute__((vector_size(16))) __device__ *' (global / AS1)
//   and the LDS-side equivalent (shared / AS3).
typedef int i32v4 __attribute__((vector_size(16)));
typedef __attribute__((address_space(1))) i32v4* gas_i32v4_ptr;
typedef __attribute__((address_space(3))) i32v4* las_i32v4_ptr;
#endif

union Frag16 {            // one 16x32 / 32x16 bf16 WMMA operand (8 VGPRs)
  v16bf          v;
  unsigned short h[16];
  u32x4          q[2];
};

// Native fp32 -> bf16 (v_cvt_pk_bf16_f32 on gfx1250), RNE.
__device__ __forceinline__ unsigned short f2bf(float f) {
  bf16_t h = (bf16_t)f;
  return __builtin_bit_cast(unsigned short, h);
}

// ---------------------------------------------------------------------------
// Kernel 0: one-shot fp32 -> bf16 weight conversion (3 x 64 x 768).
// ---------------------------------------------------------------------------
__global__ __launch_bounds__(256)
void wcvt_kernel(const float* __restrict__ wq, const float* __restrict__ wk,
                 const float* __restrict__ wv, unsigned short* __restrict__ wbf)
{
  const float* src = (blockIdx.y == 0) ? wq : (blockIdx.y == 1) ? wk : wv;
  unsigned short* dst = wbf + (size_t)blockIdx.y * DQKV * EMBED;
  const int i4 = blockIdx.x * 256 + threadIdx.x;          // 12288 float4 / matrix
  f32x4 v = *(const f32x4*)(src + (size_t)i4 * 4);
  u16x4 h;
  h.x = f2bf(v.x); h.y = f2bf(v.y); h.z = f2bf(v.z); h.w = f2bf(v.w);
  *(u16x4*)(dst + (size_t)i4 * 4) = h;
}

// ---------------------------------------------------------------------------
// Kernel 1: QKV projection.  x[16 rows x 768] staged once in LDS as bf16,
// shared by 12 waves = 3 matrices x 4 d-tiles; weights already bf16.
// Inner loop: 2 global_load_b128 + 2 ds_load_b128 + 1 wmma.
// Q is stored pre-scaled by D^-0.5 * log2(e).
// Output: q,k row-major [row][64] bf16;  v transposed [b][d][s] bf16.
// ---------------------------------------------------------------------------
__global__ __launch_bounds__(384)
void qkv_proj_kernel(const float* __restrict__ x,
                     const unsigned short* __restrict__ wbf,
                     const float* __restrict__ bq,
                     const float* __restrict__ bk,
                     const float* __restrict__ bv,
                     unsigned short* __restrict__ qo,
                     unsigned short* __restrict__ ko,
                     unsigned short* __restrict__ vto)
{
  __shared__ unsigned short xs[16 * EMBED];          // 24 KB bf16 tile
  const int tid = threadIdx.x;
  const long rowBase = (long)blockIdx.x * 16;

  // cooperative fp32->bf16 stage of the 16x768 x-tile (x read exactly once)
  const float* xsrc = x + rowBase * EMBED;
#pragma unroll
  for (int i = 0; i < 8; ++i) {
    int idx4 = tid + i * 384;                        // 3072 float4's total
    f32x4 v = *(const f32x4*)(xsrc + (size_t)idx4 * 4);
    u16x4 h;
    h.x = f2bf(v.x); h.y = f2bf(v.y); h.z = f2bf(v.z); h.w = f2bf(v.w);
    *(u16x4*)&xs[idx4 * 4] = h;
  }
  __syncthreads();

  const int lane  = tid & 31;
  const int wave  = tid >> 5;          // 0..11
  const int mat   = wave >> 2;         // 0=q 1=k 2=v
  const int dTile = wave & 3;          // 16-wide slice of D=64
  const int half  = lane >> 4;         // which K-half this lane holds
  const int rowA  = lane & 15;

  const unsigned short* Wb = wbf + (size_t)mat * DQKV * EMBED;
  const float* Bb = (mat == 0) ? bq : (mat == 1) ? bk : bv;

  v8f acc = {};
#pragma unroll 4
  for (int kb = 0; kb < EMBED; kb += 32) {
    Frag16 a, b;
    // A (16x32, x-tile) from LDS: two contiguous 16B chunks per lane
    const unsigned short* ar = &xs[rowA * EMBED];
    const int o1 = kb + (half ? 8 : 0);
    a.q[0] = *(const u32x4*)(ar + o1);
    a.q[1] = *(const u32x4*)(ar + o1 + 16);
    // B (32x16, W^T chunk): two 8-bf16 chunks (same k order as A) from the W row
    const unsigned short* wr =
        Wb + (size_t)(dTile * 16 + rowA) * EMBED + kb + (half ? 8 : 0);
    b.q[0] = *(const u32x4*)(wr);
    b.q[1] = *(const u32x4*)(wr + 16);
    acc = __builtin_amdgcn_wmma_f32_16x16x32_bf16(false, a.v, false, b.v,
                                                  (short)0, acc, false, false);
  }

  // C/D layout: lane holds col d = dTile*16+rowA, rows (half?8:0)+r
  const float bias   = Bb[dTile * 16 + rowA];
  const long  mRow0  = rowBase + (half ? 8 : 0);
  const int   dcol   = dTile * 16 + rowA;

  if (mat == 0) {                       // Q: fold softmax scale * log2(e)
    const float sc = 0.125f * 1.4426950408889634f;   // D^-0.5 * log2(e)
#pragma unroll
    for (int r = 0; r < 8; ++r)
      qo[(size_t)(mRow0 + r) * DQKV + dcol] = f2bf((acc[r] + bias) * sc);
  } else if (mat == 1) {                // K row-major
#pragma unroll
    for (int r = 0; r < 8; ++r)
      ko[(size_t)(mRow0 + r) * DQKV + dcol] = f2bf(acc[r] + bias);
  } else {                              // V transposed: vt[b][d][s]
    const int bIdx  = (int)(rowBase >> 12);
    const int sBase = (int)(mRow0 & 4095);
    u16x4 lo, hi;
    lo.x = f2bf(acc[0] + bias); lo.y = f2bf(acc[1] + bias);
    lo.z = f2bf(acc[2] + bias); lo.w = f2bf(acc[3] + bias);
    hi.x = f2bf(acc[4] + bias); hi.y = f2bf(acc[5] + bias);
    hi.z = f2bf(acc[6] + bias); hi.w = f2bf(acc[7] + bias);
    unsigned short* dst = vto + (((size_t)(bIdx * DQKV + dcol)) << 12) + sBase;
    *(u16x4*)dst       = lo;
    *(u16x4*)(dst + 4) = hi;
  }
}

// ---------------------------------------------------------------------------
// Kernel 2: flash attention. 8 waves/block, one 16-query tile per wave, all
// waves share one batch's K/V stream. Per 32-key step the block stages the
// K tile (32x64 bf16, 4KB) and V^T tile (64x32 bf16, 4KB) into LDS with
// async global->LDS b128 copies (one per thread per tile, ASYNCcnt-tracked,
// triple-buffered, one barrier per step); fragments then come from LDS.
// Per step per wave: 4 wmma (S^T), online softmax, shfl_xor(16) P^T build,
// 4 wmma (O^T += V^T x P^T).
// ---------------------------------------------------------------------------
__global__ __launch_bounds__(256)
void attn_kernel(const unsigned short* __restrict__ qm,
                 const unsigned short* __restrict__ km,
                 const unsigned short* __restrict__ vtm,
                 float* __restrict__ out)
{
  __shared__ unsigned short ldsK[3][32 * DQKV];    // 3 x 4 KB, [key][d]
  __shared__ unsigned short ldsV[3][DQKV * 32];    // 3 x 4 KB, [d][key]

  const int tid      = threadIdx.x;
  const int lane     = tid & 31;
  const int wave     = tid >> 5;
  const long qRow0   = ((long)blockIdx.x * 8 + wave) * 16;  // query row base
  const int  b       = (int)(qRow0 >> 12);
  const int  colQ    = lane & 15;                // query column / A row
  const int  half    = lane >> 4;
  const int  o8      = half ? 8 : 0;

  // Persistent Q^T B-fragments (d = 0..31 and 32..63)
  Frag16 bQ0, bQ1;
  {
    const unsigned short* qr = qm + (size_t)(qRow0 + colQ) * DQKV + (half ? 8 : 0);
    bQ0.q[0] = *(const u32x4*)(qr);
    bQ0.q[1] = *(const u32x4*)(qr + 16);
    bQ1.q[0] = *(const u32x4*)(qr + 32);
    bQ1.q[1] = *(const u32x4*)(qr + 48);
  }

  const unsigned short* kB = km  + (((size_t)b) << 12) * DQKV;
  const unsigned short* vB = vtm + (((size_t)b * DQKV) << 12);

  // per-thread staging coordinates: one b128 per thread per tile
  const int kRow = tid >> 3;                 // 0..31 (key within tile)
  const int kOff = (tid & 7) * 8;            // bf16 chunk within K row
  const int vRow = tid >> 2;                 // 0..63 (d)
  const int vOff = (tid & 3) * 8;            // bf16 chunk within V^T row

  auto issueTile = [&](int t0, int stg) {
    const unsigned short* gk = kB + (size_t)(t0 + kRow) * DQKV + kOff;
    const unsigned short* gv = vB + (((size_t)vRow) << 12) + t0 + vOff;
    unsigned short* lk = &ldsK[stg][kRow * DQKV + kOff];
    unsigned short* lv = &ldsV[stg][vRow * 32 + vOff];
#if HAVE_ASYNC_LDS
    __builtin_amdgcn_global_load_async_to_lds_b128(
        (gas_i32v4_ptr)(void*)gk, (las_i32v4_ptr)(void*)lk, 0, 0);
    __builtin_amdgcn_global_load_async_to_lds_b128(
        (gas_i32v4_ptr)(void*)gv, (las_i32v4_ptr)(void*)lv, 0, 0);
#else
    *(u32x4*)lk = *(const u32x4*)gk;         // synchronous fallback
    *(u32x4*)lv = *(const u32x4*)gv;
#endif
  };

  v8f acc[4] = {{}, {}, {}, {}};             // O^T, 4 d-tiles of 16
  float M = -1e30f, l = 0.0f;
  const v8f z = {};

  issueTile(0, 0);
  int stg = 0, stgN = 1;

  for (int ti = 0; ti < SEQ / 32; ++ti) {
    if (ti + 1 < SEQ / 32) {
      issueTile((ti + 1) * 32, stgN);        // prefetch next tile (async)
#if HAVE_ASYNC_LDS
      __builtin_amdgcn_s_wait_asynccnt(2);   // current tile's 2 ops retired
#endif
    } else {
#if HAVE_ASYNC_LDS
      __builtin_amdgcn_s_wait_asynccnt(0);
#endif
    }
    __syncthreads();                         // tile `stg` visible block-wide

    // ---- K A-fragments from LDS: rows = keys, K-dim = d ----
    const unsigned short* kb0 = &ldsK[stg][colQ * DQKV];
    const unsigned short* kb1 = kb0 + 16 * DQKV;
    Frag16 a00, a01, a10, a11;
    a00.q[0] = *(const u32x4*)(kb0 + o8);       a00.q[1] = *(const u32x4*)(kb0 + 16 + o8);
    a01.q[0] = *(const u32x4*)(kb0 + 32 + o8);  a01.q[1] = *(const u32x4*)(kb0 + 48 + o8);
    a10.q[0] = *(const u32x4*)(kb1 + o8);       a10.q[1] = *(const u32x4*)(kb1 + 16 + o8);
    a11.q[0] = *(const u32x4*)(kb1 + 32 + o8);  a11.q[1] = *(const u32x4*)(kb1 + 48 + o8);

    v8f s0 = __builtin_amdgcn_wmma_f32_16x16x32_bf16(false, a01.v, false, bQ1.v, (short)0, z,  false, false);
    s0     = __builtin_amdgcn_wmma_f32_16x16x32_bf16(false, a00.v, false, bQ0.v, (short)0, s0, false, false);
    v8f s1 = __builtin_amdgcn_wmma_f32_16x16x32_bf16(false, a11.v, false, bQ1.v, (short)0, z,  false, false);
    s1     = __builtin_amdgcn_wmma_f32_16x16x32_bf16(false, a10.v, false, bQ0.v, (short)0, s1, false, false);

    // ---- online softmax (per-query column = per-lane scalar stats) ----
    float mt = s0[0];
#pragma unroll
    for (int r = 0; r < 8; ++r) { mt = fmaxf(mt, s0[r]); mt = fmaxf(mt, s1[r]); }
    mt = fmaxf(mt, __shfl_xor(mt, 16, 32));  // other key half, same query
    const float newM  = fmaxf(M, mt);
    const float alpha = exp2f(M - newM);     // scores already in log2 domain
    M = newM;

    float p0[8], p1[8], sum = 0.0f;
#pragma unroll
    for (int r = 0; r < 8; ++r) {
      p0[r] = exp2f(s0[r] - M);
      p1[r] = exp2f(s1[r] - M);
      sum += p0[r] + p1[r];
    }
    sum += __shfl_xor(sum, 16, 32);
    l = l * alpha + sum;
#pragma unroll
    for (int dt = 0; dt < 4; ++dt)
#pragma unroll
      for (int r = 0; r < 8; ++r) acc[dt][r] *= alpha;

    // ---- build P^T B-fragment (32 keys x 16 queries) via half-exchange ----
    Frag16 bP;
#pragma unroll
    for (int r = 0; r < 8; ++r) {
      const float xp0 = __shfl_xor(p0[r], 16, 32);
      const float xp1 = __shfl_xor(p1[r], 16, 32);
      const float lo  = half ? xp1   : p0[r];  // keys (half?16:0)+r
      const float hi  = half ? p1[r] : xp0;    // keys (half?24:8)+r
      bP.h[r]     = f2bf(lo);
      bP.h[r + 8] = f2bf(hi);
    }

    // ---- O^T += V^T x P^T (V^T fragments from LDS) ----
#pragma unroll
    for (int dt = 0; dt < 4; ++dt) {
      const unsigned short* vp = &ldsV[stg][(dt * 16 + colQ) * 32];
      Frag16 aV;
      aV.q[0] = *(const u32x4*)(vp + 2 * o8);
      aV.q[1] = *(const u32x4*)(vp + 2 * o8 + 8);
      acc[dt] = __builtin_amdgcn_wmma_f32_16x16x32_bf16(false, aV.v, false, bP.v,
                                                        (short)0, acc[dt], false, false);
    }

    stg  = stgN;
    stgN = (stgN == 2) ? 0 : stgN + 1;
  }

  // ---- epilogue: normalize and store fp32 output [B,S,64] ----
  const float inv = 1.0f / l;
  const size_t outRow = (size_t)(qRow0 + colQ) * DQKV;
#pragma unroll
  for (int dt = 0; dt < 4; ++dt) {
    f32x4 oA, oB;
    oA.x = acc[dt][0] * inv; oA.y = acc[dt][1] * inv;
    oA.z = acc[dt][2] * inv; oA.w = acc[dt][3] * inv;
    oB.x = acc[dt][4] * inv; oB.y = acc[dt][5] * inv;
    oB.z = acc[dt][6] * inv; oB.w = acc[dt][7] * inv;
    float* dst = out + outRow + dt * 16 + o8;
    *(f32x4*)dst       = oA;
    *(f32x4*)(dst + 4) = oB;
  }
}

// ---------------------------------------------------------------------------
// Launcher. Workspace: Q (8MB) + K (8MB) + V^T (8MB) + W_bf16 (294KB) in d_ws.
// ---------------------------------------------------------------------------
extern "C" void kernel_launch(void* const* d_in, const int* in_sizes, int n_in,
                              void* d_out, int out_size, void* d_ws, size_t ws_size,
                              hipStream_t stream) {
  const float* x  = (const float*)d_in[0];
  const float* wq = (const float*)d_in[1];
  const float* bq = (const float*)d_in[2];
  const float* wk = (const float*)d_in[3];
  const float* bk = (const float*)d_in[4];
  const float* wv = (const float*)d_in[5];
  const float* bv = (const float*)d_in[6];
  float* out = (float*)d_out;

  const size_t rows = (size_t)NB * SEQ;           // 65536
  unsigned short* qb  = (unsigned short*)d_ws;
  unsigned short* kb  = qb + rows * DQKV;
  unsigned short* vtb = kb + rows * DQKV;
  unsigned short* wbf = vtb + rows * DQKV;

  hipLaunchKernelGGL(wcvt_kernel, dim3(48, 3), dim3(256), 0, stream,
                     wq, wk, wv, wbf);
  hipLaunchKernelGGL(qkv_proj_kernel, dim3((unsigned)(rows / 16)), dim3(384), 0, stream,
                     x, wbf, bq, bk, bv, qb, kb, vtb);
  hipLaunchKernelGGL(attn_kernel, dim3((unsigned)(rows / 16 / 8)), dim3(256), 0, stream,
                     qb, kb, vtb, out);
}
